// Attention_23811298689871
// MI455X (gfx1250) — matrix operand order repaired and checked by reading, not compiled, not run on hardware
//
#include <hip/hip_runtime.h>

typedef __attribute__((ext_vector_type(16))) _Float16 v16h;
typedef __attribute__((ext_vector_type(8)))  _Float16 v8h;
typedef __attribute__((ext_vector_type(4)))  _Float16 v4h;
typedef __attribute__((ext_vector_type(8)))  float    v8f;
typedef __attribute__((ext_vector_type(4)))  float    v4f;

#define QSCALE 0.17677669529663687f   // 32^-0.5

// ---- WMMA fragment helpers (CDNA5 wave32 layouts, CDNA5 ISA) ----
static __device__ __forceinline__ v16h cat88(v8h lo, v8h hi) {
  return __builtin_shufflevector(lo, hi, 0,1,2,3,4,5,6,7,8,9,10,11,12,13,14,15);
}
// A-operand (16x32 f16): lane holds one row; 16 contiguous K-halves
// K[base..base+15]; caller passes p = row + kbase (+16 for hi lanes)
static __device__ __forceinline__ v16h load_a(const _Float16* p) {
  return cat88(*(const v8h*)p, *(const v8h*)(p + 8));
}
// B-operand (32x16 f16): lane holds one column (= row of B^T); 16 contiguous K-halves
static __device__ __forceinline__ v16h load_b(const _Float16* p) {
  return cat88(*(const v8h*)p, *(const v8h*)(p + 8));
}
static __device__ __forceinline__ v8f wmma16(v16h a, v16h b, v8f c) {
  return __builtin_amdgcn_wmma_f32_16x16x32_f16(false, a, false, b, (short)0, c, false, false);
}

// =====================================================================
// Kernel 0: position-bias MLP -> relT in WMMA D-tile layout
// relT[((h*4+mt)*4+nt)*256 + lane*8 + j]  (coalesced b128 loads in attn)
// plus weight transpose/convert to f16
// =====================================================================
__global__ __launch_bounds__(256) void prep_kernel(
    const float* __restrict__ qkv_w, const float* __restrict__ proj_w,
    const float* __restrict__ ppw,  const float* __restrict__ ppb,
    const float* __restrict__ ln1g, const float* __restrict__ ln1b,
    const float* __restrict__ fc1w, const float* __restrict__ fc1b,
    const float* __restrict__ ln2g, const float* __restrict__ ln2b,
    const float* __restrict__ fc2w, const float* __restrict__ fc2b,
    const float* __restrict__ ln3g, const float* __restrict__ ln3b,
    const float* __restrict__ fc3w, const float* __restrict__ fc3b,
    const float* __restrict__ biases, const int* __restrict__ rel_idx,
    float* __restrict__ relT, _Float16* __restrict__ wqkvT, _Float16* __restrict__ wprojT)
{
  __shared__ float posS[225 * 8];
  const int tid = threadIdx.x;
  if (blockIdx.x == 0) {
    if (tid < 225) {
      float h[16], t[16];
      const float b0 = biases[2*tid], b1 = biases[2*tid + 1];
      for (int p = 0; p < 16; p++) h[p] = b0 * ppw[p] + b1 * ppw[16 + p] + ppb[p];
      auto lnrelu = [&](const float* in, float* out, const float* g, const float* bb) {
        float m = 0.f; for (int i = 0; i < 16; i++) m += in[i]; m *= (1.f/16.f);
        float v = 0.f; for (int i = 0; i < 16; i++) { float d = in[i]-m; v += d*d; } v *= (1.f/16.f);
        float inv = rsqrtf(v + 1e-5f);
        for (int i = 0; i < 16; i++) { float x = (in[i]-m)*inv*g[i] + bb[i]; out[i] = x > 0.f ? x : 0.f; }
      };
      lnrelu(h, t, ln1g, ln1b);
      for (int q = 0; q < 16; q++) { float s = fc1b[q]; for (int p = 0; p < 16; p++) s += t[p]*fc1w[p*16+q]; h[q] = s; }
      lnrelu(h, t, ln2g, ln2b);
      for (int q = 0; q < 16; q++) { float s = fc2b[q]; for (int p = 0; p < 16; p++) s += t[p]*fc2w[p*16+q]; h[q] = s; }
      lnrelu(h, t, ln3g, ln3b);
      for (int hd = 0; hd < 8; hd++) { float s = fc3b[hd]; for (int p = 0; p < 16; p++) s += t[p]*fc3w[p*8+hd]; posS[tid*8+hd] = s; }
    }
    __syncthreads();
    for (int t = tid; t < 8 * 4096; t += 256) {
      const int j    = t & 7;
      const int lane = (t >> 3) & 31;
      const int nt   = (t >> 8) & 3;
      const int mt   = (t >> 10) & 3;
      const int hh   = t >> 12;
      const int r = mt * 16 + j + (lane >> 4) * 8;
      const int c = nt * 16 + (lane & 15);
      relT[t] = posS[rel_idx[r * 64 + c] * 8 + hh];
    }
  }
  const int gid = blockIdx.x * 256 + tid, gsz = gridDim.x * 256;
  for (int i = gid; i < 768 * 256; i += gsz) { int n = i >> 8, k = i & 255; wqkvT[i] = (_Float16)qkv_w[k * 768 + n]; }
  for (int i = gid; i < 256 * 256; i += gsz) { int n = i >> 8, k = i & 255; wprojT[i] = (_Float16)proj_w[k * 256 + n]; }
}

// =====================================================================
// Kernel 1: per-window QKV GEMM.  One k-loop accumulates Q,K,V together
// (A fragments loaded once, 12-WMMA bursts, B pipelined one k-step ahead).
// Outputs: q[b][h][t][32] (pre-scaled), k[b][h][t][32], vt[b][h][d][64]
// =====================================================================
__global__ __launch_bounds__(256) void qkv_kernel(
    const float* __restrict__ x, const _Float16* __restrict__ wqkvT,
    const float* __restrict__ qkvb,
    _Float16* __restrict__ qg, _Float16* __restrict__ kg, _Float16* __restrict__ vtg)
{
  __shared__ _Float16 xs[64 * 264];    // x tile, pitch 264 halves
  __shared__ _Float16 sbuf[64 * 136];  // 4-head staging, pitch 136
  const int tid = threadIdx.x;
  const int b = blockIdx.x;
  { // load + convert x tile
    const int r = tid >> 2, c0 = (tid & 3) * 64;
    const float4* src = (const float4*)(x + ((size_t)b * 64 + r) * 256 + c0);
    _Float16* dst = xs + r * 264 + c0;
#pragma unroll
    for (int i = 0; i < 16; i++) {
      float4 v = src[i];
      v4h hv; hv[0]=(_Float16)v.x; hv[1]=(_Float16)v.y; hv[2]=(_Float16)v.z; hv[3]=(_Float16)v.w;
      *(v4h*)(dst + 4 * i) = hv;
    }
  }
  __syncthreads();
  const int w = tid >> 5, lane = tid & 31, l15 = lane & 15, hi = lane >> 4;
  const int hl = w >> 1;
  const v8f vzero = {0,0,0,0,0,0,0,0};
  for (int hb = 0; hb < 2; hb++) {
    const int hbase = hb * 4;
    const int col_base = (hbase + hl) * 32 + (w & 1) * 16 + l15;  // within one 256-col section
    const _Float16* btb[3];
#pragma unroll
    for (int s = 0; s < 3; s++) btb[s] = wqkvT + (size_t)(s * 256 + col_base) * 256 + hi * 16;

    v8f acc[3][4];
#pragma unroll
    for (int s = 0; s < 3; s++)
#pragma unroll
      for (int mt = 0; mt < 4; mt++) acc[s][mt] = vzero;

    v16h bm[3];
#pragma unroll
    for (int s = 0; s < 3; s++) bm[s] = load_b(btb[s]);

#pragma unroll
    for (int k0 = 0; k0 < 8; k0++) {
      v16h am[4];
#pragma unroll
      for (int mt = 0; mt < 4; mt++)
        am[mt] = load_a(xs + (mt * 16 + l15) * 264 + k0 * 32 + hi * 16);
      v16h bmn[3];
      if (k0 < 7) {
#pragma unroll
        for (int s = 0; s < 3; s++) bmn[s] = load_b(btb[s] + (k0 + 1) * 32);
      }
#pragma unroll
      for (int s = 0; s < 3; s++)
#pragma unroll
        for (int mt = 0; mt < 4; mt++)
          acc[s][mt] = wmma16(am[mt], bm[s], acc[s][mt]);
      if (k0 < 7) {
#pragma unroll
        for (int s = 0; s < 3; s++) bm[s] = bmn[s];
      }
    }

    // stage + write out, one section at a time
    const int loc = hl * 32 + (w & 1) * 16 + l15;
    for (int sec = 0; sec < 3; sec++) {
      const float bias = qkvb[sec * 256 + col_base];
      const float scale = (sec == 0) ? QSCALE : 1.0f;
#pragma unroll
      for (int mt = 0; mt < 4; mt++)
#pragma unroll
        for (int j = 0; j < 8; j++)
          sbuf[(mt * 16 + j + hi * 8) * 136 + loc] = (_Float16)((acc[sec][mt][j] + bias) * scale);
      __syncthreads();
      if (sec < 2) {  // q / k : row-major [b][h][t][32], coalesced
        _Float16* dbase = (sec == 0) ? qg : kg;
        const int h_l = tid >> 6, row = tid & 63;
        const _Float16* sp = sbuf + row * 136 + h_l * 32;
        _Float16* dp = dbase + (((size_t)b * 8 + hbase + h_l) * 64 + row) * 32;
#pragma unroll
        for (int i = 0; i < 4; i++) *(v8h*)(dp + 8 * i) = *(const v8h*)(sp + 8 * i);
      } else {        // v : transposed [b][h][d][64]
        const int half = tid >> 7, idx = tid & 127, h_l = idx >> 5, d = idx & 31;
        _Float16 tmp[32];
#pragma unroll
        for (int t2 = 0; t2 < 32; t2++) tmp[t2] = sbuf[(half * 32 + t2) * 136 + h_l * 32 + d];
        _Float16* dp = vtg + (((size_t)b * 8 + hbase + h_l) * 32 + d) * 64 + half * 32;
#pragma unroll
        for (int i = 0; i < 4; i++) {
          v8h v;
#pragma unroll
          for (int j = 0; j < 8; j++) v[j] = tmp[8 * i + j];
          *(v8h*)(dp + 8 * i) = v;
        }
      }
      __syncthreads();
    }
  }
}

// =====================================================================
// Kernel 2: attention per (window, head).  K/V B-fragments hoisted
// (mt-invariant), bias via coalesced b128 tile-layout loads,
// softmax in registers with half-wave shuffles.
// =====================================================================
__global__ __launch_bounds__(128) void attn_kernel(
    const _Float16* __restrict__ qg, const _Float16* __restrict__ kg,
    const _Float16* __restrict__ vtg, const float* __restrict__ relT,
    _Float16* __restrict__ Yg)
{
  __shared__ _Float16 psh[4 * 64 * 72];  // P per wave, pitch 72
  const int tid = threadIdx.x;
  const int b = blockIdx.x >> 1, hbase = (blockIdx.x & 1) * 4;
  const int w = tid >> 5, lane = tid & 31, l15 = lane & 15, hi = lane >> 4;
  const int h = hbase + w;
  const _Float16* qh  = qg  + ((size_t)b * 8 + h) * 2048;
  const _Float16* kh  = kg  + ((size_t)b * 8 + h) * 2048;
  const _Float16* vth = vtg + ((size_t)b * 8 + h) * 2048;
  const float* rbt = relT + (size_t)h * 4096 + lane * 8;
  _Float16* pw = psh + w * (64 * 72);
  const v8f vzero = {0,0,0,0,0,0,0,0};
  float rinv[4][8];

  // K fragments: one per nt, shared by all mt
  v16h bk[4];
#pragma unroll
  for (int nt = 0; nt < 4; nt++)
    bk[nt] = load_b(kh + (nt * 16 + l15) * 32 + hi * 16);

#pragma unroll
  for (int mt = 0; mt < 4; mt++) {
    const int m0 = mt * 16;
    v16h aq = load_a(qh + (m0 + l15) * 32 + hi * 16);
    v8f sacc[4];
#pragma unroll
    for (int nt = 0; nt < 4; nt++) sacc[nt] = wmma16(aq, bk[nt], vzero);
    // add relative-position bias (tile-layout, coalesced)
#pragma unroll
    for (int nt = 0; nt < 4; nt++) {
      const float* rp = rbt + (mt * 4 + nt) * 256;
      const v4f b0 = *(const v4f*)(rp);
      const v4f b1 = *(const v4f*)(rp + 4);
#pragma unroll
      for (int j = 0; j < 4; j++) { sacc[nt][j] += b0[j]; sacc[nt][j + 4] += b1[j]; }
    }
    const int r = m0 + hi * 8;  // + j
#pragma unroll
    for (int j = 0; j < 8; j++) {
      float m = sacc[0][j];
#pragma unroll
      for (int nt = 1; nt < 4; nt++) m = fmaxf(m, sacc[nt][j]);
      m = fmaxf(m, __shfl_xor(m, 8, 32));
      m = fmaxf(m, __shfl_xor(m, 4, 32));
      m = fmaxf(m, __shfl_xor(m, 2, 32));
      m = fmaxf(m, __shfl_xor(m, 1, 32));
      float sum = 0.f;
#pragma unroll
      for (int nt = 0; nt < 4; nt++) {
        float e = __expf(sacc[nt][j] - m);
        sacc[nt][j] = e;
        sum += e;
      }
      sum += __shfl_xor(sum, 8, 32);
      sum += __shfl_xor(sum, 4, 32);
      sum += __shfl_xor(sum, 2, 32);
      sum += __shfl_xor(sum, 1, 32);
      rinv[mt][j] = 1.0f / sum;
    }
#pragma unroll
    for (int nt = 0; nt < 4; nt++)
#pragma unroll
      for (int j = 0; j < 8; j++)
        pw[(r + j) * 72 + nt * 16 + l15] = (_Float16)sacc[nt][j];
  }

  // V fragments: [ks][dt], shared by all mt
  v16h bv[2][2];
#pragma unroll
  for (int ks = 0; ks < 2; ks++)
#pragma unroll
    for (int dt = 0; dt < 2; dt++)
      bv[ks][dt] = load_b(vth + (dt * 16 + l15) * 64 + ks * 32 + hi * 16);

#pragma unroll
  for (int mt = 0; mt < 4; mt++) {
    const int m0 = mt * 16;
    v8f oacc[2]; oacc[0] = vzero; oacc[1] = vzero;
#pragma unroll
    for (int ks = 0; ks < 2; ks++) {
      v16h ap = load_a(pw + (m0 + l15) * 72 + ks * 32 + hi * 16);
      oacc[0] = wmma16(ap, bv[ks][0], oacc[0]);
      oacc[1] = wmma16(ap, bv[ks][1], oacc[1]);
    }
    _Float16* yd = Yg + (size_t)b * 64 * 256 + h * 32;
    const int r = m0 + hi * 8;
#pragma unroll
    for (int dt = 0; dt < 2; dt++)
#pragma unroll
      for (int j = 0; j < 8; j++)
        yd[(size_t)(r + j) * 256 + dt * 16 + l15] = (_Float16)(oacc[dt][j] * rinv[mt][j]);
  }
}

// =====================================================================
// Kernel 3: per-window projection GEMM.  Y tile fetched with CDNA5
// async global->LDS copy (ASYNCcnt); f32 output + bias, coalesced stores.
// =====================================================================
__global__ __launch_bounds__(256) void proj_kernel(
    const _Float16* __restrict__ Yg, const _Float16* __restrict__ wprojT,
    const float* __restrict__ pb, float* __restrict__ out)
{
  __shared__ _Float16 ys[64 * 264];
  __shared__ float tbuf[8 * 16 * 20];
  const int tid = threadIdx.x, b = blockIdx.x;
  { // async copy Y tile (f16) straight into LDS, then wait + barrier
    const int r = tid >> 2, c0 = (tid & 3) * 64;
    const _Float16* src = Yg + ((size_t)b * 64 + r) * 256 + c0;
    _Float16* dst = ys + r * 264 + c0;
#pragma unroll
    for (int i = 0; i < 8; i++) {
      unsigned lds_off = (unsigned)(uintptr_t)(dst + 8 * i);
      const void* gp = (const void*)(src + 8 * i);
      asm volatile("global_load_async_to_lds_b128 %0, %1, off"
                   :: "v"(lds_off), "v"(gp) : "memory");
    }
    asm volatile("s_wait_asynccnt 0" ::: "memory");
  }
  __syncthreads();
  const int w = tid >> 5, lane = tid & 31, l15 = lane & 15, hi = lane >> 4;
  const v8f vzero = {0,0,0,0,0,0,0,0};
  v8f acc[4][2];
#pragma unroll
  for (int mt = 0; mt < 4; mt++) { acc[mt][0] = vzero; acc[mt][1] = vzero; }
  const _Float16* btb0 = wprojT + (size_t)(w * 32 + l15) * 256 + hi * 16;
  const _Float16* btb1 = btb0 + (size_t)16 * 256;
  v16h bm0 = load_b(btb0), bm1 = load_b(btb1);
#pragma unroll
  for (int k0 = 0; k0 < 8; k0++) {
    v16h am[4];
#pragma unroll
    for (int mt = 0; mt < 4; mt++)
      am[mt] = load_a(ys + (mt * 16 + l15) * 264 + k0 * 32 + hi * 16);
    v16h bn0, bn1;
    if (k0 < 7) { bn0 = load_b(btb0 + (k0 + 1) * 32); bn1 = load_b(btb1 + (k0 + 1) * 32); }
#pragma unroll
    for (int mt = 0; mt < 4; mt++) {
      acc[mt][0] = wmma16(am[mt], bm0, acc[mt][0]);
      acc[mt][1] = wmma16(am[mt], bm1, acc[mt][1]);
    }
    if (k0 < 7) { bm0 = bn0; bm1 = bn1; }
  }
  float* tb = tbuf + w * (16 * 20);
#pragma unroll
  for (int mt = 0; mt < 4; mt++)
#pragma unroll
    for (int dt = 0; dt < 2; dt++) {
      const int col = w * 32 + dt * 16 + l15;
      const float bias = pb[col];
#pragma unroll
      for (int j = 0; j < 8; j++) tb[(j + hi * 8) * 20 + l15] = acc[mt][dt][j] + bias;
      // same-wave LDS ops are in-order: safe to read other lanes' writes
      const int rr = l15, cp = hi * 8;
      const float4 v0 = *(const float4*)(tb + rr * 20 + cp);
      const float4 v1 = *(const float4*)(tb + rr * 20 + cp + 4);
      float* op = out + ((size_t)b * 64 + mt * 16 + rr) * 256 + w * 32 + dt * 16 + cp;
      *(float4*)op = v0;
      *(float4*)(op + 4) = v1;
    }
}

// =====================================================================
extern "C" void kernel_launch(void* const* d_in, const int* in_sizes, int n_in,
                              void* d_out, int out_size, void* d_ws, size_t ws_size,
                              hipStream_t stream) {
  const float* x      = (const float*)d_in[0];
  const float* qkv_w  = (const float*)d_in[1];
  const float* qkv_b  = (const float*)d_in[2];
  const float* proj_w = (const float*)d_in[3];
  const float* proj_b = (const float*)d_in[4];
  const float* ppw    = (const float*)d_in[5];
  const float* ppb    = (const float*)d_in[6];
  const float* ln1g   = (const float*)d_in[7];
  const float* ln1b   = (const float*)d_in[8];
  const float* fc1w   = (const float*)d_in[9];
  const float* fc1b   = (const float*)d_in[10];
  const float* ln2g   = (const float*)d_in[11];
  const float* ln2b   = (const float*)d_in[12];
  const float* fc2w   = (const float*)d_in[13];
  const float* fc2b   = (const float*)d_in[14];
  const float* ln3g   = (const float*)d_in[15];
  const float* ln3b   = (const float*)d_in[16];
  const float* fc3w   = (const float*)d_in[17];
  const float* fc3b   = (const float*)d_in[18];
  const float* biases = (const float*)d_in[19];
  const int*   relidx = (const int*)d_in[20];
  float* out = (float*)d_out;

  char* ws = (char*)d_ws;
  float*    relT   = (float*)(ws);                      // 8*4096*4       = 131072
  _Float16* wqkvT  = (_Float16*)(ws + 131072);          // 768*256*2      = 393216
  _Float16* wprojT = (_Float16*)(ws + 524288);          // 256*256*2      = 131072
  _Float16* qg     = (_Float16*)(ws + 655360);          // 2048*8*64*32*2 = 64 MiB
  _Float16* kg     = (_Float16*)(ws + 655360  + (size_t)67108864);
  _Float16* vtg    = (_Float16*)(ws + 655360  + (size_t)67108864 * 2);
  _Float16* Yg     = (_Float16*)(ws + 655360  + (size_t)67108864 * 3);

  prep_kernel<<<96, 256, 0, stream>>>(qkv_w, proj_w, ppw, ppb, ln1g, ln1b, fc1w, fc1b,
                                      ln2g, ln2b, fc2w, fc2b, ln3g, ln3b, fc3w, fc3b,
                                      biases, relidx, relT, wqkvT, wprojT);
  qkv_kernel<<<2048, 256, 0, stream>>>(x, wqkvT, qkv_b, qg, kg, vtg);
  attn_kernel<<<4096, 128, 0, stream>>>(qg, kg, vtg, relT, Yg);
  proj_kernel<<<2048, 256, 0, stream>>>(Yg, wprojT, proj_b, out);
}
